// LocalSolverCore_5660766896391
// MI455X (gfx1250) — matrix operand order repaired and checked by reading, not compiled, not run on hardware
//
#include <hip/hip_runtime.h>
#include <math.h>

// ---------------------------------------------------------------------------
// Types
// ---------------------------------------------------------------------------
typedef __bf16 bf16;
typedef __attribute__((ext_vector_type(16))) __bf16 v16bf;
typedef __attribute__((ext_vector_type(8)))  __bf16 v8bf;
typedef __attribute__((ext_vector_type(8)))  float  v8f;
typedef __attribute__((ext_vector_type(4)))  int    v4i;

#if defined(__AMDGCN__) && __has_builtin(__builtin_amdgcn_global_load_async_to_lds_b128)
#define HAVE_ASYNC_LDS 1
#else
#define HAVE_ASYNC_LDS 0
#endif

__device__ __forceinline__ void wait_async0() {
#if defined(__AMDGCN__) && __has_builtin(__builtin_amdgcn_s_wait_asynccnt)
  __builtin_amdgcn_s_wait_asynccnt(0);
#elif defined(__AMDGCN__)
  asm volatile("s_wait_asynccnt 0x0" ::: "memory");
#endif
}

#if HAVE_ASYNC_LDS
__device__ __forceinline__ void async_b128(const bf16* gsrc, bf16* ldst) {
  void* g = (void*)gsrc;  // drop const; builtin takes non-const AS1 v4i*
  __builtin_amdgcn_global_load_async_to_lds_b128(
      (__attribute__((address_space(1))) v4i*)g,
      (__attribute__((address_space(3))) v4i*)(void*)ldst, 0, 0);
}
#endif

// ---------------------------------------------------------------------------
// Problem constants (match reference)
// ---------------------------------------------------------------------------
constexpr int B_   = 2;
constexpr int T_   = 2048;
constexpr int DIN  = 4096;
constexpr int D_   = 512;
constexpr int H_   = 8;
constexpr int DH_  = 64;
constexpr int WWIN = 128;
constexpr int NM_  = 16;
constexpr int NHX  = 4;
constexpr int DHX  = 128;
constexpr int DFF  = 1024;
constexpr int MROWS = B_ * T_;  // 4096 token rows
constexpr int MH    = B_ * NM_; // 32 memory rows
constexpr float EPS = 1e-5f;

// ---------------------------------------------------------------------------
// WMMA fragment helpers (layouts per CDNA5 ISA 7.12.2, wave32)
//  A (16x32 bf16): lane L holds row M=L&15; half=L>>4;
//      elements 0..7  -> K = 8*half + e ; elements 8..15 -> K = 16+8*half+(e-8)
//  B (32x16 bf16): lane L holds col N=L&15; same K mapping as A
//  C/D (16x16 f32): lane L, vgpr r -> (M = r + 8*half, N = L&15)
// ---------------------------------------------------------------------------
__device__ __forceinline__ v16bf frag_cat(v8bf lo, v8bf hi) {
  v16bf f;
#pragma unroll
  for (int e = 0; e < 8; e++) { f[e] = lo[e]; f[8 + e] = hi[e]; }
  return f;
}
__device__ __forceinline__ v16bf frag_a(const bf16* base, int half) {
  v8bf lo = *(const v8bf*)(base + 8 * half);
  v8bf hi = *(const v8bf*)(base + 16 + 8 * half);
  return frag_cat(lo, hi);
}
__device__ __forceinline__ v16bf frag_b(const bf16* base, int half) {
  v8bf lo = *(const v8bf*)(base + 8 * half);
  v8bf hi = *(const v8bf*)(base + 16 + 8 * half);
  return frag_cat(lo, hi);
}
__device__ __forceinline__ v8f wmma_bf16(v16bf a, v16bf b, v8f c) {
  return __builtin_amdgcn_wmma_f32_16x16x32_bf16(false, a, false, b, (short)0, c,
                                                 false, false);
}

// ---------------------------------------------------------------------------
// Generic WMMA GEMM: C[M,N] = A[M,K](bf16,row-major) x Bt[N,K](bf16)^T
// Block tile 128x128, K-step 32, 8 waves (256 threads), wave tile 32x64.
// Double-buffered LDS staged via GLOBAL_LOAD_ASYNC_TO_LDS_B128 (ASYNCcnt).
// ---------------------------------------------------------------------------
constexpr int BM = 128, BN = 128, BK = 32, LDK = BK + 16;  // 96B rows

enum { EPI_F32 = 0, EPI_PROJ_IN = 1, EPI_RESID = 2, EPI_BF16 = 3, EPI_SILU = 4 };

template <int EPI>
__global__ __launch_bounds__(256) void k_gemm(
    const bf16* __restrict__ A, const bf16* __restrict__ Bt, int Mrows, int N,
    int K, float* __restrict__ out_a, float* __restrict__ out_b,
    bf16* __restrict__ out_bf) {
  __shared__ bf16 As[2][BM * LDK];
  __shared__ bf16 Bs[2][BN * LDK];

  const int tid  = threadIdx.x;
  const int lane = tid & 31;
  const int wave = tid >> 5;
  const int half = lane >> 4;
  const int l16  = lane & 15;
  const int m0 = blockIdx.y * BM;
  const int n0 = blockIdx.x * BN;
  const int wr = wave >> 1;  // 0..3 : 32-row strip
  const int wc = wave & 1;   // 0..1 : 64-col strip

  // Per-thread staging slot: 16 elems (two b128) of one 128x32 tile.
  const int si = tid * 16;
  const int sr = si >> 5;        // 0..127
  const int sc = si & 31;        // 0 or 16
  int agm = m0 + sr;
  if (agm > Mrows - 1) agm = Mrows - 1;  // clamp (garbage rows never stored)
  const bf16* aSrc = A + (size_t)agm * K + sc;
  const bf16* bSrc = Bt + (size_t)(n0 + sr) * K + sc;

  auto stage = [&](int buf, int k0) {
    bf16* ad = &As[buf][sr * LDK + sc];
    bf16* bd = &Bs[buf][sr * LDK + sc];
#if HAVE_ASYNC_LDS
    async_b128(aSrc + k0, ad);
    async_b128(aSrc + k0 + 8, ad + 8);
    async_b128(bSrc + k0, bd);
    async_b128(bSrc + k0 + 8, bd + 8);
#else
    *(v8bf*)(ad)     = *(const v8bf*)(aSrc + k0);
    *(v8bf*)(ad + 8) = *(const v8bf*)(aSrc + k0 + 8);
    *(v8bf*)(bd)     = *(const v8bf*)(bSrc + k0);
    *(v8bf*)(bd + 8) = *(const v8bf*)(bSrc + k0 + 8);
#endif
  };

  v8f acc[2][4];
#pragma unroll
  for (int a = 0; a < 2; a++)
#pragma unroll
    for (int j = 0; j < 4; j++) acc[a][j] = {};

  int buf = 0;
  stage(0, 0);
  wait_async0();
  __syncthreads();

  for (int k0 = 0;;) {
    const int nk = k0 + BK;
    if (nk < K) stage(buf ^ 1, nk);                      // overlap next tile
    if (nk + BK < K)
      __builtin_prefetch(bSrc + nk + BK, 0, 3);          // global_prefetch_b8

    v16bf af[2];
#pragma unroll
    for (int a = 0; a < 2; a++)
      af[a] = frag_a(&As[buf][(wr * 32 + 16 * a + l16) * LDK], half);
#pragma unroll
    for (int j = 0; j < 4; j++) {
      v16bf bfg = frag_b(&Bs[buf][(wc * 64 + 16 * j + l16) * LDK], half);
#pragma unroll
      for (int a = 0; a < 2; a++) acc[a][j] = wmma_bf16(af[a], bfg, acc[a][j]);
    }

    k0 = nk;
    if (k0 >= K) break;
    wait_async0();     // our async copies into buf^1 complete
    __syncthreads();   // everyone done reading buf, buf^1 visible
    buf ^= 1;
  }

#pragma unroll
  for (int a = 0; a < 2; a++) {
#pragma unroll
    for (int j = 0; j < 4; j++) {
#pragma unroll
      for (int r = 0; r < 8; r++) {
        const int m = m0 + wr * 32 + 16 * a + r + 8 * half;
        const int n = n0 + wc * 64 + 16 * j + l16;
        if (m < Mrows) {
          const float v = acc[a][j][r];
          const size_t idx = (size_t)m * N + n;
          if constexpr (EPI == EPI_F32) {
            out_a[idx] = v;
          } else if constexpr (EPI == EPI_PROJ_IN) {
            out_a[idx] = v;
            out_b[idx] = 0.1f * v;
          } else if constexpr (EPI == EPI_RESID) {
            out_a[idx] += v;
          } else if constexpr (EPI == EPI_BF16) {
            out_bf[idx] = (bf16)v;
          } else {  // SiLU
            out_bf[idx] = (bf16)(v / (1.0f + __expf(-v)));
          }
        }
      }
    }
  }
}

// ---------------------------------------------------------------------------
// LayerNorm: one wave per row. cols must be a multiple of 32.
// ---------------------------------------------------------------------------
template <bool OUT_BF16>
__global__ __launch_bounds__(256) void k_ln(void* __restrict__ out,
                                            const float* __restrict__ src,
                                            const float* __restrict__ g,
                                            const float* __restrict__ b,
                                            int rows, int cols) {
  const int wave = threadIdx.x >> 5, lane = threadIdx.x & 31;
  const int row = blockIdx.x * 8 + wave;
  if (row >= rows) return;
  const float* x = src + (size_t)row * cols;
  const int per = cols / 32;
  float s = 0.f, s2 = 0.f;
  for (int i = 0; i < per; i++) {
    const float v = x[lane + 32 * i];
    s += v;
    s2 += v * v;
  }
#pragma unroll
  for (int m = 16; m >= 1; m >>= 1) {
    s += __shfl_xor(s, m, 32);
    s2 += __shfl_xor(s2, m, 32);
  }
  const float mean = s / cols;
  const float var = s2 / cols - mean * mean;
  const float rstd = rsqrtf(var + EPS);
  for (int i = 0; i < per; i++) {
    const int c = lane + 32 * i;
    const float v = (x[c] - mean) * rstd * g[c] + b[c];
    if (OUT_BF16)
      ((bf16*)out)[(size_t)row * cols + c] = (bf16)v;
    else
      ((float*)out)[(size_t)row * cols + c] = v;
  }
}

// ---------------------------------------------------------------------------
// Banded local attention: 1 wave per 16-query tile per (b,h).
// Q,K: [B*T, D] bf16 (head-major cols). Vt: [B,H,DH,T] bf16. O: [B*T, D] bf16.
// ---------------------------------------------------------------------------
__global__ __launch_bounds__(32) void k_attn_local(const bf16* __restrict__ Q,
                                                   const bf16* __restrict__ Kb,
                                                   const bf16* __restrict__ Vt,
                                                   bf16* __restrict__ O) {
  __shared__ float p_lds[16][33];
  const int lane = threadIdx.x;
  const int half = lane >> 4, l16 = lane & 15;
  const int qt = blockIdx.x;
  const int bh = blockIdx.y;
  const int b = bh / H_, h = bh % H_;
  const int t0 = qt * 16;

  v16bf qf[2];
#pragma unroll
  for (int dk = 0; dk < 2; dk++) {
    const bf16* qp = Q + (size_t)(b * T_ + t0 + l16) * D_ + h * DH_ + 32 * dk;
    qf[dk] = frag_a(qp, half);
  }

  v8f oacc[4];
#pragma unroll
  for (int j = 0; j < 4; j++) oacc[j] = {};
  float mrow[8], lrow[8];
#pragma unroll
  for (int r = 0; r < 8; r++) { mrow[r] = -1e30f; lrow[r] = 0.f; }

  const int kbeg = (t0 - WWIN) > 0 ? (t0 - WWIN) : 0;
  const int kend = (t0 + 16 + WWIN) < T_ ? (t0 + 16 + WWIN) : T_;

  for (int ck = kbeg & ~31; ck < kend; ck += 32) {
    v8f sacc[2];
    float pv[2][8], cmax[8];
#pragma unroll
    for (int r = 0; r < 8; r++) cmax[r] = -1e30f;
#pragma unroll
    for (int s2 = 0; s2 < 2; s2++) {
      sacc[s2] = {};
      const int tk = ck + 16 * s2 + l16;
#pragma unroll
      for (int dk = 0; dk < 2; dk++) {
        const bf16* kp = Kb + (size_t)(b * T_ + tk) * D_ + h * DH_ + 32 * dk;
        sacc[s2] = wmma_bf16(qf[dk], frag_b(kp, half), sacc[s2]);
      }
#pragma unroll
      for (int r = 0; r < 8; r++) {
        const int tq = t0 + r + 8 * half;
        float v = sacc[s2][r] * 0.125f;  // 1/sqrt(64)
        int d = tq - tk;
        if (d < 0) d = -d;
        if (d > WWIN) v = -1e30f;
        pv[s2][r] = v;
        cmax[r] = fmaxf(cmax[r], v);
      }
    }
#pragma unroll
    for (int m = 8; m >= 1; m >>= 1)
#pragma unroll
      for (int r = 0; r < 8; r++)
        cmax[r] = fmaxf(cmax[r], __shfl_xor(cmax[r], m, 16));

    float alpha[8], csum[8];
#pragma unroll
    for (int r = 0; r < 8; r++) {
      const float mn = fmaxf(mrow[r], cmax[r]);
      alpha[r] = __expf(mrow[r] - mn);
      mrow[r] = mn;
      csum[r] = 0.f;
    }
#pragma unroll
    for (int s2 = 0; s2 < 2; s2++)
#pragma unroll
      for (int r = 0; r < 8; r++) {
        const float p = __expf(pv[s2][r] - mrow[r]);
        pv[s2][r] = p;
        csum[r] += p;
      }
#pragma unroll
    for (int m = 8; m >= 1; m >>= 1)
#pragma unroll
      for (int r = 0; r < 8; r++) csum[r] += __shfl_xor(csum[r], m, 16);
#pragma unroll
    for (int r = 0; r < 8; r++) lrow[r] = lrow[r] * alpha[r] + csum[r];
#pragma unroll
    for (int j = 0; j < 4; j++)
#pragma unroll
      for (int r = 0; r < 8; r++) oacc[j][r] *= alpha[r];

    // transpose P (C layout -> A layout) through LDS
#pragma unroll
    for (int s2 = 0; s2 < 2; s2++)
#pragma unroll
      for (int r = 0; r < 8; r++)
        p_lds[r + 8 * half][16 * s2 + l16] = pv[s2][r];
    __syncthreads();
    v16bf pf;
#pragma unroll
    for (int e = 0; e < 16; e++) {
      const int c = (e & 7) + 8 * half + 16 * (e >> 3);
      pf[e] = (bf16)p_lds[l16][c];
    }
    __syncthreads();

    const bf16* vbase = Vt + (size_t)(b * H_ + h) * DH_ * T_;
#pragma unroll
    for (int j = 0; j < 4; j++) {
      const bf16* vp = vbase + (size_t)(16 * j + l16) * T_ + ck;
      oacc[j] = wmma_bf16(pf, frag_b(vp, half), oacc[j]);
    }
  }

#pragma unroll
  for (int j = 0; j < 4; j++)
#pragma unroll
    for (int r = 0; r < 8; r++) {
      const int trow = t0 + r + 8 * half;
      O[(size_t)(b * T_ + trow) * D_ + h * DH_ + 16 * j + l16] =
          (bf16)(oacc[j][r] / lrow[r]);
    }
}

// ---------------------------------------------------------------------------
// Cross attention: 16 memory queries over all T keys. 1 wave per (b,head).
// Q: [B*NM, D], Kb: [B*T, D], Vt: [B,NHX,DHX,T], O: [B*NM, D].
// ---------------------------------------------------------------------------
__global__ __launch_bounds__(32) void k_attn_cross(const bf16* __restrict__ Q,
                                                   const bf16* __restrict__ Kb,
                                                   const bf16* __restrict__ Vt,
                                                   bf16* __restrict__ O) {
  __shared__ float p_lds[16][33];
  const int lane = threadIdx.x;
  const int half = lane >> 4, l16 = lane & 15;
  const int bh = blockIdx.x;
  const int b = bh / NHX, h = bh % NHX;

  v16bf qf[4];
#pragma unroll
  for (int dk = 0; dk < 4; dk++) {
    const bf16* qp = Q + (size_t)(b * NM_ + l16) * D_ + h * DHX + 32 * dk;
    qf[dk] = frag_a(qp, half);
  }

  v8f oacc[8];
#pragma unroll
  for (int j = 0; j < 8; j++) oacc[j] = {};
  float mrow[8], lrow[8];
#pragma unroll
  for (int r = 0; r < 8; r++) { mrow[r] = -1e30f; lrow[r] = 0.f; }

  for (int ck = 0; ck < T_; ck += 32) {
    v8f sacc[2];
    float pv[2][8], cmax[8];
#pragma unroll
    for (int r = 0; r < 8; r++) cmax[r] = -1e30f;
#pragma unroll
    for (int s2 = 0; s2 < 2; s2++) {
      sacc[s2] = {};
      const int tk = ck + 16 * s2 + l16;
#pragma unroll
      for (int dk = 0; dk < 4; dk++) {
        const bf16* kp = Kb + (size_t)(b * T_ + tk) * D_ + h * DHX + 32 * dk;
        sacc[s2] = wmma_bf16(qf[dk], frag_b(kp, half), sacc[s2]);
      }
#pragma unroll
      for (int r = 0; r < 8; r++) {
        const float v = sacc[s2][r] * 0.08838834764831845f;  // 1/sqrt(128)
        pv[s2][r] = v;
        cmax[r] = fmaxf(cmax[r], v);
      }
    }
#pragma unroll
    for (int m = 8; m >= 1; m >>= 1)
#pragma unroll
      for (int r = 0; r < 8; r++)
        cmax[r] = fmaxf(cmax[r], __shfl_xor(cmax[r], m, 16));

    float alpha[8], csum[8];
#pragma unroll
    for (int r = 0; r < 8; r++) {
      const float mn = fmaxf(mrow[r], cmax[r]);
      alpha[r] = __expf(mrow[r] - mn);
      mrow[r] = mn;
      csum[r] = 0.f;
    }
#pragma unroll
    for (int s2 = 0; s2 < 2; s2++)
#pragma unroll
      for (int r = 0; r < 8; r++) {
        const float p = __expf(pv[s2][r] - mrow[r]);
        pv[s2][r] = p;
        csum[r] += p;
      }
#pragma unroll
    for (int m = 8; m >= 1; m >>= 1)
#pragma unroll
      for (int r = 0; r < 8; r++) csum[r] += __shfl_xor(csum[r], m, 16);
#pragma unroll
    for (int r = 0; r < 8; r++) lrow[r] = lrow[r] * alpha[r] + csum[r];
#pragma unroll
    for (int j = 0; j < 8; j++)
#pragma unroll
      for (int r = 0; r < 8; r++) oacc[j][r] *= alpha[r];

#pragma unroll
    for (int s2 = 0; s2 < 2; s2++)
#pragma unroll
      for (int r = 0; r < 8; r++)
        p_lds[r + 8 * half][16 * s2 + l16] = pv[s2][r];
    __syncthreads();
    v16bf pf;
#pragma unroll
    for (int e = 0; e < 16; e++) {
      const int c = (e & 7) + 8 * half + 16 * (e >> 3);
      pf[e] = (bf16)p_lds[l16][c];
    }
    __syncthreads();

    const bf16* vbase = Vt + (size_t)(b * NHX + h) * DHX * T_;
#pragma unroll
    for (int j = 0; j < 8; j++) {
      const bf16* vp = vbase + (size_t)(16 * j + l16) * T_ + ck;
      oacc[j] = wmma_bf16(pf, frag_b(vp, half), oacc[j]);
    }
  }

#pragma unroll
  for (int j = 0; j < 8; j++)
#pragma unroll
    for (int r = 0; r < 8; r++) {
      const int m = r + 8 * half;
      O[(size_t)(b * NM_ + m) * D_ + h * DHX + 16 * j + l16] =
          (bf16)(oacc[j][r] / lrow[r]);
    }
}

// ---------------------------------------------------------------------------
// Small elementwise kernels
// ---------------------------------------------------------------------------
__global__ void k_add(float* __restrict__ o, const float* __restrict__ a,
                      const float* __restrict__ b, size_t n) {
  const size_t i = (size_t)blockIdx.x * blockDim.x + threadIdx.x;
  if (i < n) o[i] = a[i] + b[i];
}
__global__ void k_cvt_bf16(bf16* __restrict__ o, const float* __restrict__ a,
                           size_t n) {
  const size_t i = (size_t)blockIdx.x * blockDim.x + threadIdx.x;
  if (i < n) o[i] = (bf16)a[i];
}
// dst[n*K+k] = (bf16) src[k*N+n]   (weight [K,N] -> transposed bf16 [N,K])
__global__ void k_cvt_w_t(bf16* __restrict__ dst, const float* __restrict__ src,
                          int K, int N) {
  const size_t i = (size_t)blockIdx.x * blockDim.x + threadIdx.x;
  if (i < (size_t)K * N) {
    const int n = (int)(i / K), k = (int)(i % K);
    dst[i] = (bf16)src[(size_t)k * N + n];
  }
}
// dst[((b*NH+h)*DHp + d)*T + t] = src[(b*T+t)*D + h*DHp + d]
__global__ void k_transpose_v(bf16* __restrict__ dst,
                              const bf16* __restrict__ src, int NH, int DHp) {
  const size_t i = (size_t)blockIdx.x * blockDim.x + threadIdx.x;
  const size_t total = (size_t)B_ * NH * DHp * T_;
  if (i < total) {
    const int t = (int)(i % T_);
    const int d = (int)((i / T_) % DHp);
    const int h = (int)((i / ((size_t)T_ * DHp)) % NH);
    const int b = (int)(i / ((size_t)T_ * DHp * NH));
    dst[i] = src[(size_t)(b * T_ + t) * D_ + h * DHp + d];
  }
}
__global__ void k_init_zH(float* __restrict__ zH,
                          const float* __restrict__ h_init) {
  const size_t i = (size_t)blockIdx.x * blockDim.x + threadIdx.x;
  if (i < (size_t)MH * D_) zH[i] = h_init[i % ((size_t)NM_ * D_)];
}

// ---------------------------------------------------------------------------
// Host orchestration
// ---------------------------------------------------------------------------
static inline unsigned eblocks(size_t n) { return (unsigned)((n + 255) / 256); }

extern "C" void kernel_launch(void* const* d_in, const int* in_sizes, int n_in,
                              void* d_out, int out_size, void* d_ws,
                              size_t ws_size, hipStream_t stream) {
  const float* prompt = (const float*)d_in[0];
  const float* w_pi   = (const float*)d_in[1];
  const float* bn_g   = (const float*)d_in[2];
  const float* bn_b   = (const float*)d_in[3];
  const float* wq     = (const float*)d_in[4];
  const float* wk     = (const float*)d_in[5];
  const float* wv     = (const float*)d_in[6];
  const float* wo     = (const float*)d_in[7];
  const float* fn_g   = (const float*)d_in[8];
  const float* fn_b   = (const float*)d_in[9];
  const float* wg     = (const float*)d_in[10];
  const float* wd     = (const float*)d_in[11];
  const float* h_init = (const float*)d_in[12];
  const float* hn_g   = (const float*)d_in[13];
  const float* hn_b   = (const float*)d_in[14];
  const float* hq_w   = (const float*)d_in[15];
  const float* hk_w   = (const float*)d_in[16];
  const float* hv_w   = (const float*)d_in[17];
  const float* ho_w   = (const float*)d_in[18];
  const float* w_po   = (const float*)d_in[19];
  const float* on_g   = (const float*)d_in[20];
  const float* on_b   = (const float*)d_in[21];

  char* p = (char*)d_ws;
  auto alloc = [&](size_t bytes) -> char* {
    char* r = p;
    p += (bytes + 255) & ~(size_t)255;
    return r;
  };
  // transposed bf16 weights
  bf16* wt_pi = (bf16*)alloc((size_t)DIN * D_ * 2);
  bf16* wt_q  = (bf16*)alloc((size_t)D_ * D_ * 2);
  bf16* wt_k  = (bf16*)alloc((size_t)D_ * D_ * 2);
  bf16* wt_v  = (bf16*)alloc((size_t)D_ * D_ * 2);
  bf16* wt_o  = (bf16*)alloc((size_t)D_ * D_ * 2);
  bf16* wt_g  = (bf16*)alloc((size_t)D_ * DFF * 2);
  bf16* wt_d  = (bf16*)alloc((size_t)DFF * D_ * 2);
  bf16* wt_hq = (bf16*)alloc((size_t)D_ * D_ * 2);
  bf16* wt_hk = (bf16*)alloc((size_t)D_ * D_ * 2);
  bf16* wt_hv = (bf16*)alloc((size_t)D_ * D_ * 2);
  bf16* wt_ho = (bf16*)alloc((size_t)D_ * D_ * 2);
  bf16* wt_po = (bf16*)alloc((size_t)D_ * DIN * 2);
  // activations
  bf16*  xbf   = (bf16*)alloc((size_t)MROWS * DIN * 2);
  float* e_f   = (float*)alloc((size_t)MROWS * D_ * 4);
  float* zA    = (float*)alloc((size_t)MROWS * D_ * 4);
  float* zB    = (float*)alloc((size_t)MROWS * D_ * 4);
  bf16*  hbf   = (bf16*)alloc((size_t)MROWS * D_ * 2);
  bf16*  qbf   = (bf16*)alloc((size_t)MROWS * D_ * 2);
  bf16*  kbf   = (bf16*)alloc((size_t)MROWS * D_ * 2);
  bf16*  vbf   = (bf16*)alloc((size_t)MROWS * D_ * 2);
  bf16*  vT    = (bf16*)alloc((size_t)MROWS * D_ * 2);
  bf16*  aobf  = (bf16*)alloc((size_t)MROWS * D_ * 2);
  bf16*  gbf   = (bf16*)alloc((size_t)MROWS * DFF * 2);
  float* zH    = (float*)alloc((size_t)MH * D_ * 4);
  bf16*  hqin  = (bf16*)alloc((size_t)MH * D_ * 2);
  bf16*  hqbf  = (bf16*)alloc((size_t)MH * D_ * 2);
  bf16*  hoin  = (bf16*)alloc((size_t)MH * D_ * 2);
  bf16*  zhbf  = (bf16*)alloc((size_t)MH * D_ * 2);
  float* tmpo  = (float*)alloc((size_t)MH * DIN * 4);

  auto gemm_grid = [](int M, int N) {
    return dim3((unsigned)(N / BN), (unsigned)((M + BM - 1) / BM));
  };

  // --- weight conversion (transposed, bf16) ---
  k_cvt_w_t<<<eblocks((size_t)DIN * D_), 256, 0, stream>>>(wt_pi, w_pi, DIN, D_);
  k_cvt_w_t<<<eblocks((size_t)D_ * D_), 256, 0, stream>>>(wt_q, wq, D_, D_);
  k_cvt_w_t<<<eblocks((size_t)D_ * D_), 256, 0, stream>>>(wt_k, wk, D_, D_);
  k_cvt_w_t<<<eblocks((size_t)D_ * D_), 256, 0, stream>>>(wt_v, wv, D_, D_);
  k_cvt_w_t<<<eblocks((size_t)D_ * D_), 256, 0, stream>>>(wt_o, wo, D_, D_);
  k_cvt_w_t<<<eblocks((size_t)D_ * DFF), 256, 0, stream>>>(wt_g, wg, D_, DFF);
  k_cvt_w_t<<<eblocks((size_t)DFF * D_), 256, 0, stream>>>(wt_d, wd, DFF, D_);
  k_cvt_w_t<<<eblocks((size_t)D_ * D_), 256, 0, stream>>>(wt_hq, hq_w, D_, D_);
  k_cvt_w_t<<<eblocks((size_t)D_ * D_), 256, 0, stream>>>(wt_hk, hk_w, D_, D_);
  k_cvt_w_t<<<eblocks((size_t)D_ * D_), 256, 0, stream>>>(wt_hv, hv_w, D_, D_);
  k_cvt_w_t<<<eblocks((size_t)D_ * D_), 256, 0, stream>>>(wt_ho, ho_w, D_, D_);
  k_cvt_w_t<<<eblocks((size_t)D_ * DIN), 256, 0, stream>>>(wt_po, w_po, D_, DIN);

  // --- input projection: e = X @ W_in ; z_L = 0.1 e ---
  k_cvt_bf16<<<eblocks((size_t)MROWS * DIN), 256, 0, stream>>>(
      xbf, prompt, (size_t)MROWS * DIN);
  k_init_zH<<<eblocks((size_t)MH * D_), 256, 0, stream>>>(zH, h_init);
  k_gemm<EPI_PROJ_IN><<<gemm_grid(MROWS, D_), 256, 0, stream>>>(
      xbf, wt_pi, MROWS, D_, DIN, e_f, zA, nullptr);

  float* zcur = zA;
  float* zoth = zB;
  const size_t ND = (size_t)MROWS * D_;

  for (int s = 0; s < 3; s++) {
    for (int it = 0; it < 4; it++) {
      float* x = zoth;
      k_add<<<eblocks(ND), 256, 0, stream>>>(x, zcur, e_f, ND);
      k_ln<true><<<(MROWS + 7) / 8, 256, 0, stream>>>(hbf, x, bn_g, bn_b,
                                                      MROWS, D_);
      k_gemm<EPI_BF16><<<gemm_grid(MROWS, D_), 256, 0, stream>>>(
          hbf, wt_q, MROWS, D_, D_, nullptr, nullptr, qbf);
      k_gemm<EPI_BF16><<<gemm_grid(MROWS, D_), 256, 0, stream>>>(
          hbf, wt_k, MROWS, D_, D_, nullptr, nullptr, kbf);
      k_gemm<EPI_BF16><<<gemm_grid(MROWS, D_), 256, 0, stream>>>(
          hbf, wt_v, MROWS, D_, D_, nullptr, nullptr, vbf);
      k_transpose_v<<<eblocks(ND), 256, 0, stream>>>(vT, vbf, H_, DH_);
      k_attn_local<<<dim3(T_ / 16, B_ * H_), 32, 0, stream>>>(qbf, kbf, vT,
                                                              aobf);
      k_gemm<EPI_RESID><<<gemm_grid(MROWS, D_), 256, 0, stream>>>(
          aobf, wt_o, MROWS, D_, D_, x, nullptr, nullptr);
      k_ln<true><<<(MROWS + 7) / 8, 256, 0, stream>>>(hbf, x, fn_g, fn_b,
                                                      MROWS, D_);
      k_gemm<EPI_SILU><<<gemm_grid(MROWS, DFF), 256, 0, stream>>>(
          hbf, wt_g, MROWS, DFF, D_, nullptr, nullptr, gbf);
      k_gemm<EPI_RESID><<<gemm_grid(MROWS, D_), 256, 0, stream>>>(
          gbf, wt_d, MROWS, D_, DFF, x, nullptr, nullptr);
      float* t = zcur; zcur = zoth; zoth = t;  // z_L := x
    }
    // --- cross attention: z_H update ---
    k_ln<true><<<(MH + 7) / 8, 256, 0, stream>>>(hqin, zH, hn_g, hn_b, MH, D_);
    k_gemm<EPI_BF16><<<gemm_grid(MH, D_), 256, 0, stream>>>(
        hqin, wt_hq, MH, D_, D_, nullptr, nullptr, hqbf);
    k_cvt_bf16<<<eblocks(ND), 256, 0, stream>>>(hbf, zcur, ND);
    k_gemm<EPI_BF16><<<gemm_grid(MROWS, D_), 256, 0, stream>>>(
        hbf, wt_hk, MROWS, D_, D_, nullptr, nullptr, kbf);
    k_gemm<EPI_BF16><<<gemm_grid(MROWS, D_), 256, 0, stream>>>(
        hbf, wt_hv, MROWS, D_, D_, nullptr, nullptr, vbf);
    k_transpose_v<<<eblocks(ND), 256, 0, stream>>>(vT, vbf, NHX, DHX);
    k_attn_cross<<<dim3(B_ * NHX), 32, 0, stream>>>(hqbf, kbf, vT, hoin);
    k_gemm<EPI_RESID><<<gemm_grid(MH, D_), 256, 0, stream>>>(
        hoin, wt_ho, MH, D_, D_, zH, nullptr, nullptr);
  }

  // --- output: LN(z_H @ W_out) ---
  k_cvt_bf16<<<eblocks((size_t)MH * D_), 256, 0, stream>>>(zhbf, zH,
                                                           (size_t)MH * D_);
  k_gemm<EPI_F32><<<gemm_grid(MH, DIN), 256, 0, stream>>>(
      zhbf, wt_po, MH, DIN, D_, tmpo, nullptr, nullptr);
  k_ln<false><<<(MH + 7) / 8, 256, 0, stream>>>((void*)d_out, tmpo, on_g, on_b,
                                                MH, DIN);
}
